// MyCRF_83846351552863
// MI455X (gfx1250) — matrix operand order repaired and checked by reading, not compiled, not run on hardware
//
#include <hip/hip_runtime.h>

typedef _Float16 v16h __attribute__((ext_vector_type(16)));
typedef _Float16 v8h  __attribute__((ext_vector_type(8)));
typedef float    v8f  __attribute__((ext_vector_type(8)));
typedef float    v4f  __attribute__((ext_vector_type(4)));

constexpr int kBS = 1024;
constexpr int kSL = 512;
constexpr int kT  = 64;
constexpr int kPD = 8;      // prefetch distance (steps ahead)

// compiler scheduling fence; in-wave LDS ordering is guaranteed by the
// in-order DS pipeline, this just stops compiler code motion
__device__ __forceinline__ void wave_fence() {
    __builtin_amdgcn_wave_barrier();
    asm volatile("" ::: "memory");
}

__global__ void crf_init_ws(float* ws) { ws[0] = 0.0f; ws[1] = 0.0f; }
__global__ void crf_finalize(const float* ws, float* out) { out[0] = ws[0] / ws[1]; }

__global__ __launch_bounds__(32)
void crf_forward_wmma(const float* __restrict__ e,
                      const int* __restrict__ tags,
                      const unsigned char* __restrict__ mask,
                      const float* __restrict__ st,
                      const float* __restrict__ et,
                      const float* __restrict__ t,
                      float* __restrict__ ws)
{
    // LDS: transpose buffers only. alpha is carried as (q, L):
    //   alpha[m][k] = L[m] + log(q[m][k]),  q normalized so max_k q = 1.
    __shared__ __align__(16) float    dS[16][68];   // raw S = q x expT, row layout
    __shared__ __align__(16) _Float16 pS[16][72];   // q as f16 (WMMA A staging)

    const int lane = threadIdx.x;       // single wave per block
    const int r    = lane & 15;
    const int h    = lane >> 4;
    const int b0   = blockIdx.x * 16;
    const int b_row = b0 + r;           // lane pair (r, r+16) owns batch row b_row

    // ---------------- B fragments: expT = exp(t), WMMA B layout, loop-invariant
    v16h Bf[2][4];
    for (int kh = 0; kh < 2; ++kh) {
        for (int nt = 0; nt < 4; ++nt) {
            const int n  = nt * 16 + r;
            const int ks = kh * 32 + h * 8;
            v16h tmp;
            #pragma unroll
            for (int i = 0; i < 16; ++i)
                tmp[i] = (_Float16)__expf(t[(ks + (i & 7) + (i >> 3) * 16) * kT + n]);
            Bf[kh][nt] = tmp;
        }
    }

    // ---------------- init: alpha0 = st + e[:,0,:]  ->  (q, L)
    float qRow[32];
    float L;
    {
        float a0[32];
        float mx0 = -1e30f;
        const float* e0 = e + (size_t)b_row * kSL * kT + h * 32;
        #pragma unroll
        for (int q = 0; q < 8; ++q) {
            v4f ev = *(const v4f*)(e0 + q * 4);
            #pragma unroll
            for (int j = 0; j < 4; ++j) {
                const float a = st[h * 32 + q * 4 + j] + ev[j];
                a0[q * 4 + j] = a;
                mx0 = fmaxf(mx0, a);
            }
        }
        const float mx = fmaxf(mx0, __shfl_xor(mx0, 16, 32));
        L = mx;
        #pragma unroll
        for (int k = 0; k < 32; ++k)
            qRow[k] = __expf(a0[k] - mx);
    }

    // ---------------- gold-path score (halves duplicate; only h==0 accumulates)
    int   tag_prev = tags[(size_t)b_row * kSL];
    float score    = st[tag_prev] + e[(size_t)b_row * kSL * kT + tag_prev];
    float cnt      = (float)mask[(size_t)b_row * kSL];

    for (int s = 1; s < kSL; ++s) {
        // ---- warm WGP$ for the dependent chain (uniform branch)
        if (s + kPD < kSL)
            __builtin_prefetch(e + ((size_t)b_row * kSL + (s + kPD)) * kT + h * 32, 0, 3);

        // ---- stage q as f16 into pS (contiguous, 4x b128 stores)
        {
            _Float16* dst = &pS[r][h * 32];
            #pragma unroll
            for (int q = 0; q < 4; ++q) {
                v8h v;
                #pragma unroll
                for (int j = 0; j < 8; ++j)
                    v[j] = (_Float16)qRow[q * 8 + j];
                *(v8h*)(dst + q * 8) = v;
            }
        }
        wave_fence();

        // ---- A fragments (16x32 f16 A layout): half0 K{0..7,16..23}, half1 +8
        const int kc = h * 8;
        const v8h a00 = *(const v8h*)&pS[r][kc];
        const v8h a01 = *(const v8h*)&pS[r][kc + 16];
        const v8h a10 = *(const v8h*)&pS[r][kc + 32];
        const v8h a11 = *(const v8h*)&pS[r][kc + 48];
        v16h A0, A1;
        #pragma unroll
        for (int i = 0; i < 8; ++i) {
            A0[i] = a00[i]; A0[i + 8] = a01[i];
            A1[i] = a10[i]; A1[i + 8] = a11[i];
        }

        // ---- S = q x expT via 8 WMMAs (full EXEC)
        v8f C[4];
        #pragma unroll
        for (int nt = 0; nt < 4; ++nt) {
            v8f z = {};
            z = __builtin_amdgcn_wmma_f32_16x16x32_f16(false, A0, false, Bf[0][nt],
                                                       (short)0, z, false, false);
            C[nt] = __builtin_amdgcn_wmma_f32_16x16x32_f16(false, A1, false, Bf[1][nt],
                                                           (short)0, z, false, false);
        }

        // ---- gold-path score update
        {
            const int   tg = tags[(size_t)b_row * kSL + s];
            const float mf = (float)mask[(size_t)b_row * kSL + s];
            score += (t[tag_prev * kT + tg] +
                      e[((size_t)b_row * kSL + s) * kT + tg]) * mf;
            cnt += mf;
            tag_prev = tg;
        }

        // ---- transpose: scatter raw S[m,n] into dS row layout (no log here!)
        #pragma unroll
        for (int nt = 0; nt < 4; ++nt) {
            const int n = nt * 16 + r;
            #pragma unroll
            for (int v = 0; v < 8; ++v)
                dS[v + h * 8][n] = C[nt][v];
        }
        wave_fence();

        // ---- row-lane update: u = S * exp(e), renormalize, ONE log per row
        {
            const float* er = e + ((size_t)b_row * kSL + s) * kT + h * 32;
            const float  mf = (float)mask[(size_t)b_row * kSL + s];
            float u[32];
            float mxLoc = -1e30f;
            #pragma unroll
            for (int q = 0; q < 8; ++q) {
                v4f dv = *(const v4f*)&dS[r][h * 32 + q * 4];
                v4f ev = *(const v4f*)(er + q * 4);
                #pragma unroll
                for (int j = 0; j < 4; ++j) {
                    const float uu = dv[j] * __expf(ev[j]);
                    u[q * 4 + j] = uu;
                    mxLoc = fmaxf(mxLoc, uu);
                }
            }
            const float mx = fmaxf(mxLoc, __shfl_xor(mxLoc, 16, 32));
            const float rv = __builtin_amdgcn_rcpf(mx);
            const bool  upd = (mf != 0.0f);
            L = upd ? (L + __logf(mx)) : L;
            #pragma unroll
            for (int k = 0; k < 32; ++k)
                qRow[k] = upd ? (u[k] * rv) : qRow[k];
        }
        wave_fence();
    }

    // ---------------- logZ per row: L + log(sum q * exp(et))
    float sm = 0.0f;
    #pragma unroll
    for (int k = 0; k < 32; ++k)
        sm += qRow[k] * __expf(et[h * 32 + k]);
    sm += __shfl_xor(sm, 16, 32);
    const float logZ = L + __logf(sm);

    // ---------------- end-tag score + global accumulation (half 0 only)
    if (h == 0) {
        const int last  = (int)cnt - 1;
        const int tlast = tags[(size_t)b_row * kSL + last];
        atomicAdd(&ws[0], (score + et[tlast]) - logZ);
        atomicAdd(&ws[1], cnt);
    }
}

extern "C" void kernel_launch(void* const* d_in, const int* in_sizes, int n_in,
                              void* d_out, int out_size, void* d_ws, size_t ws_size,
                              hipStream_t stream) {
    const float*         e    = (const float*)d_in[0];
    const int*           tags = (const int*)d_in[1];
    const unsigned char* mask = (const unsigned char*)d_in[2];
    const float*         st   = (const float*)d_in[3];
    const float*         et   = (const float*)d_in[4];
    const float*         t    = (const float*)d_in[5];
    float* out = (float*)d_out;
    float* ws  = (float*)d_ws;

    crf_init_ws<<<1, 1, 0, stream>>>(ws);
    crf_forward_wmma<<<kBS / 16, 32, 0, stream>>>(e, tags, mask, st, et, t, ws);
    crf_finalize<<<1, 1, 0, stream>>>(ws, out);
}
